// Retain_12343736008955
// MI455X (gfx1250) — matrix operand order repaired and checked by reading, not compiled, not run on hardware
//
#include <hip/hip_runtime.h>
#include <hip/hip_bf16.h>

typedef __attribute__((ext_vector_type(16))) _Float16 v16h;
typedef __attribute__((ext_vector_type(8)))  _Float16 v8h;
typedef __attribute__((ext_vector_type(4)))  _Float16 v4h;
typedef __attribute__((ext_vector_type(8)))  float    v8f;

#define TDIM  64
#define BDIM  64
#define MROWS 4096            // T*B
#define EDIM  128
#define HDIM  128
#define G3    384             // 3*H
#define DIN   4096
#define DOUT  4096

// WMMA GEMM tiling
#define BM  128
#define BN  128
#define BK  32
#define LDK 40                // padded LDS row stride (halves); 80B keeps 16B align

__device__ __forceinline__ float sigf(float v) { return 1.0f / (1.0f + __expf(-v)); }

// B fragment 32x16 from W^T tile laid [n][k]: lane col = nt*16 + l15; halves split K at 8/24
__device__ __forceinline__ v16h load_bfrag(const _Float16* base, int nt, int l15, int hf)
{
  const _Float16* pb = base + (nt * 16 + l15) * LDK;
  v8h lo = *(const v8h*)(pb + hf * 8);
  v8h hi = *(const v8h*)(pb + 16 + hf * 8);
  return __builtin_shufflevector(lo, hi, 0,1,2,3,4,5,6,7,8,9,10,11,12,13,14,15);
}

// C[M,N] = epilogue( (aScale*A[M,K]) @ W[N,K]^T + bias[N] )
// A, W row-major f32; converted to f16 in LDS; f32 WMMA accumulate.
// Double-buffered LDS; global loads for chunk c+1 overlap WMMA of chunk c.
// All 8 B fragments are loaded into live registers before the WMMA burst so the
// wave pays one s_wait_dscnt per chunk instead of one per tile.
// act: 0 = none, 1 = sigmoid, 2 = tanh,
//      3 = fused context accumulate: C[m,:] += alphaS[m]*tanh(val)*emb[j(m),:]  (N==EDIM)
__global__ __launch_bounds__(256) void retain_wmma_gemm(
    const float* __restrict__ A, const float* __restrict__ W,
    const float* __restrict__ bias, float* __restrict__ C,
    int M, int N, int K, float aScale, int act,
    const float* __restrict__ emb, const float* __restrict__ alphaS, int step)
{
  __shared__ __align__(16) _Float16 sA[2][BM * LDK];
  __shared__ __align__(16) _Float16 sB[2][BN * LDK];

  const int tid  = threadIdx.x;
  const int lane = tid & 31;
  const int wave = tid >> 5;          // 0..7 : wave's 16-row slice of the 128-row tile
  const int hf   = lane >> 4;         // half-wave select
  const int l15  = lane & 15;

  const int m0 = blockIdx.y * BM;
  const int n0 = blockIdx.x * BN;

  // staging index for this thread: 4 float4 slots per array per chunk
  const int srow0 = tid >> 3;               // + it*32
  const int skk   = (tid & 7) << 2;

  v8f acc[8];
  #pragma unroll
  for (int t = 0; t < 8; ++t)
    #pragma unroll
    for (int e = 0; e < 8; ++e) acc[t][e] = 0.0f;

  const int nc = K / BK;
  float4 fA[4], fW[4];

  // prologue: issue global loads for chunk 0 into registers
  #pragma unroll
  for (int it = 0; it < 4; ++it) {
    const int row = srow0 + it * 32;
    fA[it] = *(const float4*)(A + (size_t)(m0 + row) * K + skk);
    fW[it] = *(const float4*)(W + (size_t)(n0 + row) * K + skk);
  }

  for (int c = 0; c < nc; ++c) {
    const int buf = c & 1;

    // convert staged registers to f16 and store into LDS buffer `buf`
    #pragma unroll
    for (int it = 0; it < 4; ++it) {
      const int row = srow0 + it * 32;
      v4h h4;
      h4[0] = (_Float16)(fA[it].x * aScale); h4[1] = (_Float16)(fA[it].y * aScale);
      h4[2] = (_Float16)(fA[it].z * aScale); h4[3] = (_Float16)(fA[it].w * aScale);
      *(v4h*)(sA[buf] + row * LDK + skk) = h4;
      v4h g4;
      g4[0] = (_Float16)fW[it].x; g4[1] = (_Float16)fW[it].y;
      g4[2] = (_Float16)fW[it].z; g4[3] = (_Float16)fW[it].w;
      *(v4h*)(sB[buf] + row * LDK + skk) = g4;
    }
    __syncthreads();

    // issue next chunk's global loads; they complete while we do WMMA below
    if (c + 1 < nc) {
      const int k0 = (c + 1) * BK;
      #pragma unroll
      for (int it = 0; it < 4; ++it) {
        const int row = srow0 + it * 32;
        fA[it] = *(const float4*)(A + (size_t)(m0 + row) * K + k0 + skk);
        fW[it] = *(const float4*)(W + (size_t)(n0 + row) * K + k0 + skk);
      }
    }

    // A fragment 16x32: lane row = wave*16 + l15; halves split K at 8/24
    const _Float16* pa = sA[buf] + (wave * 16 + l15) * LDK;
    v8h alo = *(const v8h*)(pa + hf * 8);         // K = hf*8 + 0..7
    v8h ahi = *(const v8h*)(pa + 16 + hf * 8);    // K = 16 + hf*8 + 0..7
    v16h af = __builtin_shufflevector(alo, ahi, 0,1,2,3,4,5,6,7,8,9,10,11,12,13,14,15);

    // load ALL B fragments into distinct live registers (16x ds_load_b128),
    // then issue the 8 WMMAs back-to-back behind a single dscnt wait.
    v16h bf[8];
    #pragma unroll
    for (int nt = 0; nt < 8; ++nt) bf[nt] = load_bfrag(sB[buf], nt, l15, hf);
    #pragma unroll
    for (int nt = 0; nt < 8; ++nt)
      acc[nt] = __builtin_amdgcn_wmma_f32_16x16x32_f16(
          false, af, false, bf[nt], (short)0, acc[nt], false, false);
    // no trailing barrier needed: next iteration stores to the *other* buffer,
    // and the barrier at its top orders store-vs-load on that buffer.
  }

  // epilogue: C/D layout — VGPR v, lanes 0-15: (M=v, N=l15); lanes 16-31: (M=8+v)
  if (act == 3) {
    #pragma unroll
    for (int nt = 0; nt < 8; ++nt) {
      const int col = n0 + nt * 16 + l15;
      const float bv = bias[col];
      #pragma unroll
      for (int v = 0; v < 8; ++v) {
        const int row = m0 + wave * 16 + hf * 8 + v;
        const int i = row >> 6, b = row & 63;
        const int j = i - step;
        if (j >= 0) {
          const float val = tanhf(acc[nt][v] + bv);
          C[(size_t)row * N + col] +=
              alphaS[row] * val * emb[(size_t)((j << 6) + b) * EDIM + col];
        }
      }
    }
  } else {
    #pragma unroll
    for (int nt = 0; nt < 8; ++nt) {
      const int col = n0 + nt * 16 + l15;
      const float bv = bias ? bias[col] : 0.0f;
      #pragma unroll
      for (int v = 0; v < 8; ++v) {
        const int row = m0 + wave * 16 + hf * 8 + v;
        float val = acc[nt][v] + bv;
        if (act == 1)      val = sigf(val);
        else if (act == 2) val = tanhf(val);
        C[(size_t)row * N + col] = val;
      }
    }
  }
}

// One block per row m=(i,b); 128 threads over hidden dim.
// GI rows gathered at j = clip(i-s, 0); GH already holds b_hh; computes GRU
// update and (scan A only) the alpha logit L[m] = 0.5*dot(h_new, w_alpha)+b_alpha.
__global__ __launch_bounds__(128) void retain_gru_gate(
    const float* __restrict__ GI, const float* __restrict__ GH,
    const float* __restrict__ h_in, float* __restrict__ h_out,
    const float* __restrict__ w_alpha, const float* __restrict__ b_alpha,
    float* __restrict__ Lrow, int s)
{
  __shared__ float red[128];
  const int m = blockIdx.x;
  const int i = m >> 6, b = m & 63;
  int j = i - s; if (j < 0) j = 0;                // clip (unused states masked later)
  const int mi = (j << 6) + b;
  const int h = threadIdx.x;

  const float r  = sigf(GI[mi * G3 + h]           + GH[m * G3 + h]);
  const float z  = sigf(GI[mi * G3 + HDIM + h]    + GH[m * G3 + HDIM + h]);
  const float n  = tanhf(GI[mi * G3 + 2*HDIM + h] + r * GH[m * G3 + 2*HDIM + h]);
  const float hn = (1.0f - z) * n + z * h_in[m * HDIM + h];
  h_out[m * HDIM + h] = hn;

  if (Lrow) {
    red[h] = hn * w_alpha[h];
    __syncthreads();
    for (int off = 64; off > 0; off >>= 1) {
      if (h < off) red[h] += red[h + off];
      __syncthreads();
    }
    if (h == 0) Lrow[m] = 0.5f * red[0] + b_alpha[0];
  }
}

// alpha over valid prefix j<=i, folded with 1/(i+1) mean.  A2[s][m] = alpha(j=i-s)/(i+1)
__global__ void retain_alpha_softmax(const float* __restrict__ L, float* __restrict__ A2)
{
  const int m = blockIdx.x * blockDim.x + threadIdx.x;
  if (m >= MROWS) return;
  const int i = m >> 6;
  float mx = -1e30f;
  for (int s = 0; s <= i; ++s) mx = fmaxf(mx, L[s * MROWS + m]);
  float sum = 0.0f;
  for (int s = 0; s <= i; ++s) sum += __expf(L[s * MROWS + m] - mx);
  const float inv = 1.0f / (sum * (float)(i + 1));
  for (int s = 0; s <= i; ++s) A2[s * MROWS + m] = __expf(L[s * MROWS + m] - mx) * inv;
}

__global__ void retain_zero_f32(float* __restrict__ p, int n)
{
  int i = blockIdx.x * 256 + threadIdx.x;
  if (i < n) p[i] = 0.0f;
}

extern "C" void kernel_launch(void* const* d_in, const int* in_sizes, int n_in,
                              void* d_out, int out_size, void* d_ws, size_t ws_size,
                              hipStream_t stream) {
  (void)in_sizes; (void)n_in; (void)out_size; (void)ws_size;
  const float* x       = (const float*)d_in[0];
  const float* W_emb   = (const float*)d_in[1];
  const float* b_emb   = (const float*)d_in[2];
  const float* W_ih_a  = (const float*)d_in[3];
  const float* W_hh_a  = (const float*)d_in[4];
  const float* b_ih_a  = (const float*)d_in[5];
  const float* b_hh_a  = (const float*)d_in[6];
  const float* W_ih_b  = (const float*)d_in[7];
  const float* W_hh_b  = (const float*)d_in[8];
  const float* b_ih_b  = (const float*)d_in[9];
  const float* b_hh_b  = (const float*)d_in[10];
  const float* w_alpha = (const float*)d_in[11];
  const float* b_alpha = (const float*)d_in[12];
  const float* W_beta  = (const float*)d_in[13];
  const float* b_beta  = (const float*)d_in[14];
  const float* W_out   = (const float*)d_in[15];
  const float* b_out   = (const float*)d_in[16];
  float* out = (float*)d_out;

  // workspace carve-up (f32), ~28 MB total
  float* ws  = (float*)d_ws;
  float* emb = ws;  ws += (size_t)MROWS * EDIM;     // 4096x128
  float* GA  = ws;  ws += (size_t)MROWS * G3;       // 4096x384
  float* GB  = ws;  ws += (size_t)MROWS * G3;
  float* GH  = ws;  ws += (size_t)MROWS * G3;
  float* h0  = ws;  ws += (size_t)MROWS * HDIM;
  float* h1  = ws;  ws += (size_t)MROWS * HDIM;
  float* L   = ws;  ws += (size_t)TDIM * MROWS;     // logits [s][m]
  float* A2  = ws;  ws += (size_t)TDIM * MROWS;     // alpha/(i+1) [s][m]
  float* cbuf= ws;  ws += (size_t)MROWS * EDIM;     // context accumulator

  const dim3 blk(256);
  const float* nf = nullptr;

  // 1) emb = x @ W_emb^T + b_emb          (M=4096, N=128, K=4096)
  retain_wmma_gemm<<<dim3(EDIM / BN, MROWS / BM), blk, 0, stream>>>(
      x, W_emb, b_emb, emb, MROWS, EDIM, DIN, 1.0f, 0, nf, nf, 0);

  // 2) input-gate precompute for both GRUs (M=4096, N=384, K=128)
  retain_wmma_gemm<<<dim3(G3 / BN, MROWS / BM), blk, 0, stream>>>(
      emb, W_ih_a, b_ih_a, GA, MROWS, G3, EDIM, 1.0f, 0, nf, nf, 0);
  retain_wmma_gemm<<<dim3(G3 / BN, MROWS / BM), blk, 0, stream>>>(
      emb, W_ih_b, b_ih_b, GB, MROWS, G3, EDIM, 1.0f, 0, nf, nf, 0);

  // zero h and context accumulator
  retain_zero_f32<<<(MROWS * HDIM + 255) / 256, blk, 0, stream>>>(h0, MROWS * HDIM);
  retain_zero_f32<<<(MROWS * EDIM + 255) / 256, blk, 0, stream>>>(cbuf, MROWS * EDIM);

  // 3) scan A: 64 sequential GRU steps, emitting alpha logits
  float* hc = h0; float* hn = h1;
  for (int s = 0; s < TDIM; ++s) {
    retain_wmma_gemm<<<dim3(G3 / BN, MROWS / BM), blk, 0, stream>>>(
        hc, W_hh_a, b_hh_a, GH, MROWS, G3, HDIM, 1.0f, 0, nf, nf, 0);
    retain_gru_gate<<<MROWS, 128, 0, stream>>>(
        GA, GH, hc, hn, w_alpha, b_alpha, L + (size_t)s * MROWS, s);
    float* t = hc; hc = hn; hn = t;
  }

  // 4) alpha softmax (with 1/(i+1) mean folded in)
  retain_alpha_softmax<<<(MROWS + 255) / 256, blk, 0, stream>>>(L, A2);

  // 5) scan B with fused beta-projection + context accumulation
  retain_zero_f32<<<(MROWS * HDIM + 255) / 256, blk, 0, stream>>>(hc, MROWS * HDIM);
  for (int s = 0; s < TDIM; ++s) {
    retain_wmma_gemm<<<dim3(G3 / BN, MROWS / BM), blk, 0, stream>>>(
        hc, W_hh_b, b_hh_b, GH, MROWS, G3, HDIM, 1.0f, 0, nf, nf, 0);
    retain_gru_gate<<<MROWS, 128, 0, stream>>>(
        GB, GH, hc, hn, w_alpha, b_alpha, (float*)nullptr, s);
    // cbuf[m,:] += alpha[s,m] * tanh(0.5*h_new @ W_beta^T + b_beta) * emb[j,:]
    retain_wmma_gemm<<<dim3(EDIM / BN, MROWS / BM), blk, 0, stream>>>(
        hn, W_beta, b_beta, cbuf, MROWS, EDIM, HDIM, 0.5f, 3,
        emb, A2 + (size_t)s * MROWS, s);
    float* t = hc; hc = hn; hn = t;
  }

  // 6) out = sigmoid(c @ W_out^T + b_out)   (M=4096, N=4096, K=128)
  retain_wmma_gemm<<<dim3(DOUT / BN, MROWS / BM), blk, 0, stream>>>(
      cbuf, W_out, b_out, out, MROWS, DOUT, EDIM, 1.0f, 1, nf, nf, 0);
}
